// Attention_44401371906281
// MI455X (gfx1250) — compile-verified, matrix operand order checked by reading, not run on hardware
//
#include <hip/hip_runtime.h>
#include <hip/hip_bf16.h>

// ---------------------------------------------------------------------------
// x(16,256,640) -> N=4096 tokens; embed=inner=512; TOPK=64; alpha=0.99.
// Split-bf16 WMMA GEMMs (hi/lo, 3 WMMAs per product) for q/k, dots, and the
// Richardson solve X <- X + (xf - A X).  A (split bf16, 64 MB) is L2-resident
// (192 MB L2).  Solve GEMM: B operand staged to LDS with
// global_load_async_to_lds_b128 (ASYNCcnt double-buffering), A fragment
// reused across a 16x64 wave strip (12 WMMAs per 32-K chunk per wave).
// ---------------------------------------------------------------------------

#define N_TOK   4096
#define D_FULL  640
#define D_EMB   512
#define TOPK_N  64
#define EPS_C   2.2204460492503131e-16f
#define SOLVE_ITERS 12

typedef __attribute__((ext_vector_type(16))) __bf16 v16bf;
typedef __attribute__((ext_vector_type(8)))  __bf16 v8bf;
typedef __attribute__((ext_vector_type(8)))  float  v8f;

// ---- bf16 bit helpers ------------------------------------------------------
__device__ __forceinline__ unsigned short f2bf_rne(float f) {
  unsigned u = __float_as_uint(f);
  unsigned r = u + 0x7FFFu + ((u >> 16) & 1u);
  return (unsigned short)(r >> 16);
}
__device__ __forceinline__ float bf2f(unsigned short b) {
  return __uint_as_float(((unsigned)b) << 16);
}
__device__ __forceinline__ void split2(float f, unsigned short* h, unsigned short* l) {
  unsigned short hb = f2bf_rne(f);
  *h = hb;
  *l = f2bf_rne(f - bf2f(hb));
}
__device__ __forceinline__ unsigned ordf(float f) {
  unsigned u = __float_as_uint(f);
  return (u & 0x80000000u) ? ~u : (u | 0x80000000u);
}

// ---- WMMA fragment helpers -------------------------------------------------
// A-frag (16x32 MxK): lane L -> row (L&15); K base +8 for lanes 16..31;
// halves 0..7 at K+0..7, halves 8..15 at K+16..23.  B-frag identical pattern
// when operand is stored N-major.
__device__ __forceinline__ v16bf load_frag(const unsigned short* __restrict__ p,
                                           int ld, int r0, int k0) {
  int lane = threadIdx.x & 31;
  const unsigned short* q =
      p + (size_t)(r0 + (lane & 15)) * (size_t)ld + (size_t)(k0 + ((lane >> 4) << 3));
  v8bf a = *(const v8bf*)q;
  v8bf b = *(const v8bf*)(q + 16);
  return __builtin_shufflevector(a, b, 0,1,2,3,4,5,6,7,8,9,10,11,12,13,14,15);
}
// same pattern but from an LDS row of 32 halves (row stride = 32 elements)
__device__ __forceinline__ v16bf lds_frag(const unsigned short* base, int lane) {
  const unsigned short* q = base + ((lane >> 4) << 3);
  v8bf a = *(const v8bf*)q;
  v8bf b = *(const v8bf*)(q + 16);
  return __builtin_shufflevector(a, b, 0,1,2,3,4,5,6,7,8,9,10,11,12,13,14,15);
}

__device__ __forceinline__ v8f wmma3(v16bf ah, v16bf al, v16bf bh, v16bf bl, v8f acc) {
  acc = __builtin_amdgcn_wmma_f32_16x16x32_bf16(false, ah, false, bh, (short)0, acc, false, false);
  acc = __builtin_amdgcn_wmma_f32_16x16x32_bf16(false, ah, false, bl, (short)0, acc, false, false);
  acc = __builtin_amdgcn_wmma_f32_16x16x32_bf16(false, al, false, bh, (short)0, acc, false, false);
  return acc;
}

// ---- gfx1250 async global->LDS copy (ASYNCcnt) -----------------------------
__device__ __forceinline__ void async_copy_b128(unsigned lds_off, const void* gptr) {
  asm volatile("global_load_async_to_lds_b128 %0, %1, off"
               :: "v"(lds_off), "v"((unsigned long long)(uintptr_t)gptr)
               : "memory");
}
__device__ __forceinline__ void wait_asynccnt0() {
  asm volatile("s_wait_asynccnt 0x0" ::: "memory");
}

// ---------------------------------------------------------------------------
// 1) prep_x: x -> xe hi/lo (drop last 128 cols) and xft (640x4096 transpose)
__global__ void __launch_bounds__(256) prep_x_kernel(
    const float* __restrict__ x, unsigned short* __restrict__ xh,
    unsigned short* __restrict__ xl, float* __restrict__ xft) {
  int idx = blockIdx.x * 256 + threadIdx.x;
  int i = idx / D_FULL, d = idx - i * D_FULL;
  float v = x[idx];
  xft[(size_t)d * N_TOK + i] = v;
  if (d < D_EMB) split2(v, &xh[(size_t)i * D_EMB + d], &xl[(size_t)i * D_EMB + d]);
}

// 2) prep_w: W(512x512) -> transposed bf16 splits (n-major)
__global__ void __launch_bounds__(256) prep_w_kernel(
    const float* __restrict__ W, unsigned short* __restrict__ th,
    unsigned short* __restrict__ tl) {
  int idx = blockIdx.x * 256 + threadIdx.x;
  int kk = idx >> 9, n = idx & 511;
  split2(W[idx], &th[(size_t)n * D_EMB + kk], &tl[(size_t)n * D_EMB + kk]);
}

// 3) q/k GEMM: out(4096x512) = xe @ W, split-bf16 output
__global__ void __launch_bounds__(256) gemm_qk_kernel(
    const unsigned short* __restrict__ xh, const unsigned short* __restrict__ xl,
    const unsigned short* __restrict__ wth, const unsigned short* __restrict__ wtl,
    unsigned short* __restrict__ oh, unsigned short* __restrict__ ol) {
  int wave = threadIdx.x >> 5, lane = threadIdx.x & 31;
  int tile = blockIdx.x * 8 + wave;
  int tm = tile >> 5, tn = tile & 31;
  v8f acc = {};
#pragma unroll 2
  for (int k = 0; k < D_EMB; k += 32) {
    v16bf ah = load_frag(xh, D_EMB, tm * 16, k);
    v16bf al = load_frag(xl, D_EMB, tm * 16, k);
    v16bf bh = load_frag(wth, D_EMB, tn * 16, k);
    v16bf bl = load_frag(wtl, D_EMB, tn * 16, k);
    acc = wmma3(ah, al, bh, bl, acc);
  }
  int mrow = tm * 16 + ((lane >> 4) << 3);
  int ncol = tn * 16 + (lane & 15);
#pragma unroll
  for (int v = 0; v < 8; ++v) {
    size_t idx = (size_t)(mrow + v) * D_EMB + ncol;
    split2(acc[v], &oh[idx], &ol[idx]);
  }
}

// 4) row norms
__global__ void __launch_bounds__(256) rownorm_kernel(
    const unsigned short* __restrict__ h, const unsigned short* __restrict__ l,
    float* __restrict__ outn) {
  int row = blockIdx.x * 256 + threadIdx.x;
  const unsigned short* ph = h + (size_t)row * D_EMB;
  const unsigned short* pl = l + (size_t)row * D_EMB;
  float s = 0.f;
  for (int j = 0; j < D_EMB; ++j) {
    float v = bf2f(ph[j]) + bf2f(pl[j]);
    s += v * v;
  }
  outn[row] = s;
}

// 5) dots GEMM: dots[i][j] = (q2[i]+k2[j]-2*q_i.k_j)/512
__global__ void __launch_bounds__(256) dots_kernel(
    const unsigned short* __restrict__ qh, const unsigned short* __restrict__ ql,
    const unsigned short* __restrict__ kh, const unsigned short* __restrict__ kl,
    const float* __restrict__ qn, const float* __restrict__ kn,
    float* __restrict__ dots) {
  int wave = threadIdx.x >> 5, lane = threadIdx.x & 31;
  int tile = blockIdx.x * 8 + wave;
  int tm = tile >> 8, tn = tile & 255;
  v8f acc = {};
#pragma unroll 2
  for (int k = 0; k < D_EMB; k += 32) {
    v16bf ah = load_frag(qh, D_EMB, tm * 16, k);
    v16bf al = load_frag(ql, D_EMB, tm * 16, k);
    v16bf bh = load_frag(kh, D_EMB, tn * 16, k);
    v16bf bl = load_frag(kl, D_EMB, tn * 16, k);
    acc = wmma3(ah, al, bh, bl, acc);
  }
  int mrow = tm * 16 + ((lane >> 4) << 3);
  int ncol = tn * 16 + (lane & 15);
  float kv = kn[ncol];
  const float invD = 1.0f / (float)D_EMB;
#pragma unroll
  for (int v = 0; v < 8; ++v) {
    int m = mrow + v;
    dots[(size_t)m * N_TOK + ncol] = (qn[m] + kv - 2.0f * acc[v]) * invD;
  }
}

// 6) per-row exact top-64 threshold (32-step bitwise radix select, row in LDS)
__global__ void __launch_bounds__(256) topk_thresh_kernel(
    const float* __restrict__ dots, unsigned* __restrict__ tu) {
  __shared__ unsigned su[N_TOK];
  __shared__ int scnt;
  int row = blockIdx.x;
  const float* dr = dots + (size_t)row * N_TOK;
  for (int j = threadIdx.x; j < N_TOK; j += 256) su[j] = ordf(dr[j]);
  __syncthreads();
  unsigned t = 0;
  for (int b = 31; b >= 0; --b) {
    unsigned cand = t | (1u << b);
    if (threadIdx.x == 0) scnt = 0;
    __syncthreads();
    int c = 0;
    for (int j = threadIdx.x; j < N_TOK; j += 256) c += (su[j] >= cand) ? 1 : 0;
    atomicAdd(&scnt, c);
    __syncthreads();
    if (scnt >= TOPK_N) t = cand;
    __syncthreads();
  }
  if (threadIdx.x == 0) tu[row] = t;
}

// 7) assemble A = I - alpha*(dots*mask) + EPS, split-bf16
__global__ void __launch_bounds__(256) assemble_A_kernel(
    const float* __restrict__ dots, const unsigned* __restrict__ tu,
    const float* __restrict__ alpha_p,
    unsigned short* __restrict__ Ah, unsigned short* __restrict__ Al) {
  unsigned idx = blockIdx.x * 256u + threadIdx.x;
  unsigned i = idx >> 12, j = idx & 4095u;
  float dij = dots[(size_t)i * N_TOK + j];
  float dji = dots[(size_t)j * N_TOK + i];
  bool m = (ordf(dij) >= tu[i]) || (ordf(dji) >= tu[j]);
  float alpha = alpha_p[0];
  float a = ((i == j) ? 1.0f : 0.0f) - alpha * (m ? dij : 0.0f) + EPS_C;
  split2(a, &Ah[idx], &Al[idx]);
}

// 8) X0 = xf (transposed layout) + splits
__global__ void __launch_bounds__(256) init_x_kernel(
    const float* __restrict__ xft, float* __restrict__ Xt,
    unsigned short* __restrict__ Xh, unsigned short* __restrict__ Xl) {
  int idx = blockIdx.x * 256 + threadIdx.x;
  float v = xft[idx];
  Xt[idx] = v;
  split2(v, &Xh[idx], &Xl[idx]);
}

// 9) Yt = (A @ X)^T.  Block = 8 waves -> 128M x 64N macro-tile; wave computes
//    a 16M x 64N strip (4 accumulators) reusing its A fragment 4x.  B strip
//    (64 rows x 32 K halves, hi+lo) staged in LDS via async global->LDS
//    copies, double buffered on ASYNCcnt.
#define SB_N 64
__global__ void __launch_bounds__(256) solve_gemm_kernel(
    const unsigned short* __restrict__ Ah, const unsigned short* __restrict__ Al,
    const unsigned short* __restrict__ Xh, const unsigned short* __restrict__ Xl,
    float* __restrict__ Yt) {
  __shared__ unsigned short sb[2][2][SB_N][32];   // [buf][hi/lo][n][k] : 16 KB
  int wave = threadIdx.x >> 5, lane = threadIdx.x & 31;
  int bm = blockIdx.x / 10, bn = blockIdx.x % 10;   // 32 x 10 blocks
  int m0 = bm * 128 + wave * 16;
  int n0 = bn * SB_N;

  // async-stage assignments: thread t fills quarter (t&3) of row (t>>2)
  int nl = threadIdx.x >> 2;
  int qd = threadIdx.x & 3;
  const unsigned short* gxh = Xh + (size_t)(n0 + nl) * N_TOK + qd * 8;
  const unsigned short* gxl = Xl + (size_t)(n0 + nl) * N_TOK + qd * 8;

  async_copy_b128((unsigned)(uintptr_t)&sb[0][0][nl][qd * 8], gxh);
  async_copy_b128((unsigned)(uintptr_t)&sb[0][1][nl][qd * 8], gxl);
  wait_asynccnt0();
  __syncthreads();

  v8f acc[4] = {{}, {}, {}, {}};
  const size_t arow = (size_t)(m0 + (lane & 15)) * N_TOK;

  int buf = 0;
  for (int k = 0; k < N_TOK; k += 32) {
    int kn2 = k + 32;
    if (kn2 < N_TOK) {   // prefetch next chunk into the idle buffer
      async_copy_b128((unsigned)(uintptr_t)&sb[buf ^ 1][0][nl][qd * 8], gxh + kn2);
      async_copy_b128((unsigned)(uintptr_t)&sb[buf ^ 1][1][nl][qd * 8], gxl + kn2);
    }
    if (k + 128 < N_TOK) {
      __builtin_prefetch(Ah + arow + k + 128, 0, 1);   // global_prefetch_b8
      __builtin_prefetch(Al + arow + k + 128, 0, 1);
    }
    v16bf ah = load_frag(Ah, N_TOK, m0, k);
    v16bf al = load_frag(Al, N_TOK, m0, k);
#pragma unroll
    for (int j = 0; j < 4; ++j) {
      int nrow = j * 16 + (lane & 15);
      v16bf bh = lds_frag(&sb[buf][0][nrow][0], lane);
      v16bf bl = lds_frag(&sb[buf][1][nrow][0], lane);
      acc[j] = wmma3(ah, al, bh, bl, acc[j]);
    }
    wait_asynccnt0();
    __syncthreads();
    buf ^= 1;
  }

  int mrow = m0 + ((lane >> 4) << 3);
#pragma unroll
  for (int j = 0; j < 4; ++j) {
    int ncol = n0 + j * 16 + (lane & 15);
#pragma unroll
    for (int v = 0; v < 8; ++v)
      Yt[(size_t)ncol * N_TOK + (mrow + v)] = acc[j][v];
  }
}

// 10) Richardson update: Xt += (xft - Yt); refresh splits
__global__ void __launch_bounds__(256) update_x_kernel(
    float* __restrict__ Xt, const float* __restrict__ xft,
    const float* __restrict__ Yt, unsigned short* __restrict__ Xh,
    unsigned short* __restrict__ Xl) {
  int idx = blockIdx.x * 256 + threadIdx.x;
  float v = Xt[idx] + (xft[idx] - Yt[idx]);
  Xt[idx] = v;
  split2(v, &Xh[idx], &Xl[idx]);
}

// 11) write out row-major (b,n,d)
__global__ void __launch_bounds__(256) writeout_kernel(
    const float* __restrict__ Xt, float* __restrict__ out) {
  int idx = blockIdx.x * 256 + threadIdx.x;
  int i = idx / D_FULL, d = idx - i * D_FULL;
  out[idx] = Xt[(size_t)d * N_TOK + i];
}

// ---------------------------------------------------------------------------
extern "C" void kernel_launch(void* const* d_in, const int* in_sizes, int n_in,
                              void* d_out, int out_size, void* d_ws, size_t ws_size,
                              hipStream_t stream) {
  const float* x     = (const float*)d_in[0];
  const float* Wq    = (const float*)d_in[1];
  const float* Wk    = (const float*)d_in[2];
  const float* alpha = (const float*)d_in[3];
  float* out = (float*)d_out;

  char* ws = (char*)d_ws;
  size_t o = 0;
  auto carve = [&](size_t bytes) -> char* {
    char* p = ws + o;
    o = (o + bytes + 255) & ~(size_t)255;
    return p;
  };
  const size_t NE = (size_t)N_TOK * D_EMB, NN = (size_t)N_TOK * N_TOK,
               ND = (size_t)N_TOK * D_FULL, EE = (size_t)D_EMB * D_EMB;
  unsigned short* xe_h = (unsigned short*)carve(NE * 2);
  unsigned short* xe_l = (unsigned short*)carve(NE * 2);
  unsigned short* wqt_h = (unsigned short*)carve(EE * 2);
  unsigned short* wqt_l = (unsigned short*)carve(EE * 2);
  unsigned short* wkt_h = (unsigned short*)carve(EE * 2);
  unsigned short* wkt_l = (unsigned short*)carve(EE * 2);
  unsigned short* q_h = (unsigned short*)carve(NE * 2);
  unsigned short* q_l = (unsigned short*)carve(NE * 2);
  unsigned short* k_h = (unsigned short*)carve(NE * 2);
  unsigned short* k_l = (unsigned short*)carve(NE * 2);
  float* qn = (float*)carve(N_TOK * 4);
  float* kn = (float*)carve(N_TOK * 4);
  float* xft = (float*)carve(ND * 4);
  float* dots = (float*)carve(NN * 4);
  unsigned* tu = (unsigned*)carve(N_TOK * 4);
  unsigned short* A_h = (unsigned short*)carve(NN * 2);
  unsigned short* A_l = (unsigned short*)carve(NN * 2);
  float* Xt = (float*)carve(ND * 4);
  unsigned short* X_h = (unsigned short*)carve(ND * 2);
  unsigned short* X_l = (unsigned short*)carve(ND * 2);
  float* Yt = (float*)carve(ND * 4);
  (void)ws_size; (void)in_sizes; (void)n_in; (void)out_size;

  dim3 blk(256);
  prep_x_kernel<<<dim3(ND / 256), blk, 0, stream>>>(x, xe_h, xe_l, xft);
  prep_w_kernel<<<dim3(EE / 256), blk, 0, stream>>>(Wq, wqt_h, wqt_l);
  prep_w_kernel<<<dim3(EE / 256), blk, 0, stream>>>(Wk, wkt_h, wkt_l);
  gemm_qk_kernel<<<dim3((256 * 32) / 8), blk, 0, stream>>>(xe_h, xe_l, wqt_h, wqt_l, q_h, q_l);
  gemm_qk_kernel<<<dim3((256 * 32) / 8), blk, 0, stream>>>(xe_h, xe_l, wkt_h, wkt_l, k_h, k_l);
  rownorm_kernel<<<dim3(N_TOK / 256), blk, 0, stream>>>(q_h, q_l, qn);
  rownorm_kernel<<<dim3(N_TOK / 256), blk, 0, stream>>>(k_h, k_l, kn);
  dots_kernel<<<dim3((256 * 256) / 8), blk, 0, stream>>>(q_h, q_l, k_h, k_l, qn, kn, dots);
  topk_thresh_kernel<<<dim3(N_TOK), blk, 0, stream>>>(dots, tu);
  assemble_A_kernel<<<dim3(NN / 256), blk, 0, stream>>>(dots, tu, alpha, A_h, A_l);
  init_x_kernel<<<dim3(ND / 256), blk, 0, stream>>>(xft, Xt, X_h, X_l);
  for (int it = 0; it < SOLVE_ITERS; ++it) {
    solve_gemm_kernel<<<dim3(32 * 10), blk, 0, stream>>>(A_h, A_l, X_h, X_l, Yt);
    update_x_kernel<<<dim3(ND / 256), blk, 0, stream>>>(Xt, xft, Yt, X_h, X_l);
  }
  writeout_kernel<<<dim3(ND / 256), blk, 0, stream>>>(Xt, out);
}
